// LSTM_56745107915059
// MI455X (gfx1250) — matrix operand order repaired and checked by reading, not compiled, not run on hardware
//
#include <hip/hip_runtime.h>

// ---------------------------------------------------------------------------
// LSTM scan for MI455X (gfx1250, wave32, WMMA + Tensor Data Mover).
// L=512, B=64, E=512, H=1024.  Single persistent kernel runs all 512 steps:
//   z = [x_t | h_{t-1}] @ W + b  (bf16 WMMA, f32 accum), gates, state update,
// with a device-wide barrier between steps.  c-state lives in VGPRs.
// ---------------------------------------------------------------------------

#define L_SEQ 512
#define BATCH 64
#define EDIM  512
#define HDIM  1024
#define NWG   32          // persistent workgroups (1 per WGP, 224 KB LDS each)

typedef __attribute__((ext_vector_type(16))) __bf16 v16bf;
typedef __attribute__((ext_vector_type(8)))  float  v8f;
typedef __attribute__((ext_vector_type(4)))  unsigned int v4u;
typedef __attribute__((ext_vector_type(8)))  int v8i;
typedef __attribute__((ext_vector_type(4)))  int v4i;

union FragB16 { uint4 q[2]; v16bf v; };   // 16 bf16 = 32 bytes = 8 VGPRs/lane

__device__ __forceinline__ unsigned short f32_to_bf16_rne(float f) {
    unsigned int u = __float_as_uint(f);
    u += 0x7FFFu + ((u >> 16) & 1u);      // round-to-nearest-even
    return (unsigned short)(u >> 16);
}

__device__ __forceinline__ float fsigmoid(float x) {
    return 1.0f / (1.0f + __expf(-x));
}

// ---------------------------------------------------------------------------
// Tensor Data Mover: 2-D bf16 tile load (rows x dim0, row stride stride0
// elements) from global into LDS at byte offset lds_off.  D# per ISA ch.8.
// ---------------------------------------------------------------------------
__device__ __forceinline__ void tdm_load_2d_bf16(
    unsigned lds_off, const void* gptr,
    unsigned dim0, unsigned rows, unsigned stride0)
{
    unsigned long long ga = (unsigned long long)(uintptr_t)gptr;
    v4u g0;
    g0[0] = 1u;                                    // count=1, user descriptor
    g0[1] = lds_off;                               // lds_addr (bytes)
    g0[2] = (unsigned)ga;                          // global_addr[31:0]
    g0[3] = (unsigned)((ga >> 32) & 0x01FFFFFFu)   // global_addr[56:32]
            | (2u << 30);                          // type = 2 ("image")
    v8i g1;
    g1[0] = (int)(1u << 16);        // workgroup_mask=0, data_size=1 (2 bytes)
    g1[1] = (int)(dim0 << 16);      // tensor_dim0[15:0] (hi bits zero)
    g1[2] = (int)(rows << 16);      // tensor_dim0[31:16]=0 | tensor_dim1[15:0]
    g1[3] = (int)(dim0 << 16);      // tensor_dim1[31:16]=0 | tile_dim0
    g1[4] = (int)rows;              // tile_dim1 | tile_dim2=0
    g1[5] = (int)stride0;           // tensor_dim0_stride[31:0]
    g1[6] = 0;                      // stride0[47:32] | tensor_dim1_stride lo
    g1[7] = 0;                      // tensor_dim1_stride hi
    v4i z4 = {0, 0, 0, 0};
#if defined(__clang_major__) && (__clang_major__ >= 23)
    v8i z8 = {0, 0, 0, 0, 0, 0, 0, 0};
    __builtin_amdgcn_tensor_load_to_lds(g0, g1, z4, z4, z8, 0);
#else
    __builtin_amdgcn_tensor_load_to_lds(g0, g1, z4, z4, 0);
#endif
}

// ---------------------------------------------------------------------------
// One-time weight packing into WMMA B-fragment order.
// Packed layout: [gate][jt (H/16)][kt][lane(32)][16 bf16]; a lane's 16 bf16
// are column (jt*16 + lane&15), K = kt*32 + (lane>>4)*16 + 0..15 — the
// contiguous k order (k = 16h + s), the same order the A fragments use.
// ---------------------------------------------------------------------------
__global__ void __launch_bounds__(256) pack_wx_kernel(
    const float* __restrict__ Wf, const float* __restrict__ Wi,
    const float* __restrict__ Wg, const float* __restrict__ Wo,
    unsigned short* __restrict__ dst)
{
    unsigned idx  = blockIdx.x * 256u + threadIdx.x;   // 4*64*16*32 = 131072
    unsigned lane = idx & 31u;
    unsigned kt   = (idx >> 5) & 15u;
    unsigned jt   = (idx >> 9) & 63u;
    unsigned gate = idx >> 15;
    const float* W = (gate == 0) ? Wf : (gate == 1) ? Wi : (gate == 2) ? Wg : Wo;
    unsigned j  = jt * 16u + (lane & 15u);
    unsigned k0 = kt * 32u + (lane >> 4) * 16u;
    union { uint4 q[2]; unsigned short s[16]; } buf;
#pragma unroll
    for (int e = 0; e < 16; ++e)
        buf.s[e] = f32_to_bf16_rne(W[(size_t)(k0 + e) * HDIM + j]);
    uint4* out = (uint4*)(dst + (size_t)idx * 16u);
    out[0] = buf.q[0];
    out[1] = buf.q[1];
}

__global__ void __launch_bounds__(256) pack_wh_kernel(
    const float* __restrict__ Wf, const float* __restrict__ Wi,
    const float* __restrict__ Wg, const float* __restrict__ Wo,
    unsigned short* __restrict__ dst)
{
    unsigned idx  = blockIdx.x * 256u + threadIdx.x;   // 4*64*32*32 = 262144
    unsigned lane = idx & 31u;
    unsigned kth  = (idx >> 5) & 31u;
    unsigned jt   = (idx >> 10) & 63u;
    unsigned gate = idx >> 16;
    const float* W = (gate == 0) ? Wf : (gate == 1) ? Wi : (gate == 2) ? Wg : Wo;
    unsigned j  = jt * 16u + (lane & 15u);
    unsigned k0 = EDIM + kth * 32u + (lane >> 4) * 16u;
    union { uint4 q[2]; unsigned short s[16]; } buf;
#pragma unroll
    for (int e = 0; e < 16; ++e)
        buf.s[e] = f32_to_bf16_rne(W[(size_t)(k0 + e) * HDIM + j]);
    uint4* out = (uint4*)(dst + (size_t)idx * 16u);
    out[0] = buf.q[0];
    out[1] = buf.q[1];
}

// One-time f32 -> bf16 conversion of the whole input sequence (fully parallel).
__global__ void __launch_bounds__(256) conv_x_kernel(
    const float* __restrict__ x, unsigned short* __restrict__ xb)
{
    size_t i = ((size_t)blockIdx.x * 256u + threadIdx.x) * 8u;
    float4 f0 = *(const float4*)(x + i);
    float4 f1 = *(const float4*)(x + i + 4);
    union { uint4 q; unsigned short s[8]; } pk;
    pk.s[0] = f32_to_bf16_rne(f0.x); pk.s[1] = f32_to_bf16_rne(f0.y);
    pk.s[2] = f32_to_bf16_rne(f0.z); pk.s[3] = f32_to_bf16_rne(f0.w);
    pk.s[4] = f32_to_bf16_rne(f1.x); pk.s[5] = f32_to_bf16_rne(f1.y);
    pk.s[6] = f32_to_bf16_rne(f1.z); pk.s[7] = f32_to_bf16_rne(f1.w);
    *(uint4*)(xb + i) = pk.q;
}

__global__ void __launch_bounds__(256) init_kernel(
    const float* __restrict__ h_in,
    const float* __restrict__ bf_, const float* __restrict__ bi_,
    const float* __restrict__ bg_, const float* __restrict__ bo_,
    unsigned short* __restrict__ h0, float* __restrict__ bias,
    unsigned* __restrict__ bar)
{
    unsigned i = blockIdx.x * 256u + threadIdx.x;      // 65536 threads
    h0[i] = f32_to_bf16_rne(h_in[i]);
    if (i < 4096u) {
        unsigned g = i >> 10, j = i & 1023u;
        const float* b = (g == 0) ? bf_ : (g == 1) ? bi_ : (g == 2) ? bg_ : bo_;
        bias[i] = b[j];
    }
    if (i == 0) *bar = 0u;                             // reset grid barrier
}

// ---------------------------------------------------------------------------
// Device-wide barrier for NWG co-resident workgroups (monotonic counter).
// ---------------------------------------------------------------------------
__device__ __forceinline__ void grid_barrier(unsigned* bar, unsigned target) {
    __threadfence();
    __syncthreads();
    if (threadIdx.x == 0) {
        __hip_atomic_fetch_add(bar, 1u, __ATOMIC_RELEASE,
                               __HIP_MEMORY_SCOPE_AGENT);
        while (__hip_atomic_load(bar, __ATOMIC_ACQUIRE,
                                 __HIP_MEMORY_SCOPE_AGENT) < target)
            __builtin_amdgcn_s_sleep(1);
    }
    __syncthreads();
}

// ---------------------------------------------------------------------------
// Persistent scan kernel.  Grid: NWG=32 WGs x 256 threads (8 wave32).
// WG b owns H-columns [32b, 32b+32) for all 4 gates, for the whole scan;
// its c-state block (64x32) stays in VGPRs (8 floats/thread).
// Per step: TDM-load full A operand (x-tile 64 KB + h-tile 128 KB) into LDS,
// barrier-free K loop of v_wmma_f32_16x16x32_bf16, fused gates, grid barrier.
// ---------------------------------------------------------------------------
__global__ void __launch_bounds__(256) lstm_scan_kernel(
    const unsigned short* __restrict__ xb,       // bf16 (L, B, E)
    const unsigned short* __restrict__ WxP,      // packed bf16 x-weights
    const unsigned short* __restrict__ WhP,      // packed bf16 h-weights
    const float* __restrict__ bias,              // (4096,)
    const float* __restrict__ c_in,              // (B, H) initial c
    unsigned short* __restrict__ hA,             // bf16 (B, H) ping
    unsigned short* __restrict__ hB,             // bf16 (B, H) pong
    float* __restrict__ c_full,                  // (L, B, H)
    float* __restrict__ h_full,                  // (L, B, H)
    unsigned* __restrict__ bar)                  // grid barrier counter
{
    __shared__ __align__(16) unsigned short smAx[BATCH * EDIM];  //  64 KB
    __shared__ __align__(16) unsigned short smAh[BATCH * HDIM];  // 128 KB
    __shared__ float smZ[4 * BATCH * 32];                        //  32 KB

    const unsigned tid  = threadIdx.x;
    const unsigned wave = tid >> 5;
    const unsigned lane = tid & 31u;
    const unsigned gate = wave >> 1;                      // 0..3 (f,i,g,o)
    const unsigned jtg  = blockIdx.x * 2u + (wave & 1u);  // H 16-col tile id
    const unsigned m    = lane & 15u;
    const unsigned sel  = lane >> 4;                      // K-half select
    const unsigned j0   = blockIdx.x * 32u;

    const unsigned offAx = (unsigned)(uintptr_t)(void*)smAx;
    const unsigned offAh = (unsigned)(uintptr_t)(void*)smAh;

    // Loop-invariant: packed-B bases (uint4 units) and bias value.
    const size_t colBlk = (size_t)(gate * 64u + jtg);
    const uint4* bx = (const uint4*)WxP + colBlk * 16u * 64u + lane * 2u;
    const uint4* bh = (const uint4*)WhP + colBlk * 32u * 64u + lane * 2u;
    const float bval = bias[gate * 1024u + jtg * 16u + m];
    const unsigned clocal = (wave & 1u) * 16u + m;

    // c-state block for this WG lives in registers for the whole scan.
    float creg[8];
#pragma unroll
    for (int e = 0; e < 8; ++e) {
        unsigned idx = (unsigned)e * 256u + tid;
        creg[e] = c_in[(size_t)(idx >> 5) * HDIM + j0 + (idx & 31u)];
    }

    for (int t = 0; t < L_SEQ; ++t) {
        const unsigned short* hp = (t & 1) ? hB : hA;
        unsigned short*       hn = (t & 1) ? hA : hB;
        const unsigned short* xt = xb + (size_t)t * BATCH * EDIM;

        // Stage the whole A matrix via the TDM (wave 0; EXEC ignored,
        // tracked per-wave with TENSORcnt).
        if (wave == 0) {
            tdm_load_2d_bf16(offAx, xt, EDIM, BATCH, EDIM);
            tdm_load_2d_bf16(offAh, hp, HDIM, BATCH, HDIM);
            __builtin_amdgcn_s_wait_tensorcnt(0);
        }
        __syncthreads();

        v8f acc[4];
#pragma unroll
        for (int r = 0; r < 4; ++r) acc[r] = (v8f)0.0f;

        // ---- K loop, x part (16 slices of 32), barrier-free ----
        for (int kt = 0; kt < 16; ++kt) {
            FragB16 bfrag;
            const uint4* bp = bx + (size_t)kt * 64u;
            bfrag.q[0] = bp[0];
            bfrag.q[1] = bp[1];
#pragma unroll
            for (int r = 0; r < 4; ++r) {
                const unsigned arow = (unsigned)r * 16u + m;
                const unsigned base = arow * (unsigned)EDIM
                                      + (unsigned)kt * 32u + sel * 16u;
                FragB16 afrag;
                afrag.q[0] = *(const uint4*)(smAx + base);
                afrag.q[1] = *(const uint4*)(smAx + base + 8u);
                acc[r] = __builtin_amdgcn_wmma_f32_16x16x32_bf16(
                             false, afrag.v, false, bfrag.v,
                             (short)0, acc[r], false, false);
            }
        }
        // ---- K loop, h part (32 slices of 32) ----
        for (int kt = 0; kt < 32; ++kt) {
            FragB16 bfrag;
            const uint4* bp = bh + (size_t)kt * 64u;
            bfrag.q[0] = bp[0];
            bfrag.q[1] = bp[1];
#pragma unroll
            for (int r = 0; r < 4; ++r) {
                const unsigned arow = (unsigned)r * 16u + m;
                const unsigned base = arow * (unsigned)HDIM
                                      + (unsigned)kt * 32u + sel * 16u;
                FragB16 afrag;
                afrag.q[0] = *(const uint4*)(smAh + base);
                afrag.q[1] = *(const uint4*)(smAh + base + 8u);
                acc[r] = __builtin_amdgcn_wmma_f32_16x16x32_bf16(
                             false, afrag.v, false, bfrag.v,
                             (short)0, acc[r], false, false);
            }
        }

        // Scatter z (+bias) into LDS so all 4 gates per (row,col) meet.
#pragma unroll
        for (int r = 0; r < 4; ++r) {
#pragma unroll
            for (int v = 0; v < 8; ++v) {
                unsigned row = (unsigned)r * 16u + (unsigned)v + sel * 8u;
                smZ[(gate * 64u + row) * 32u + clocal] = acc[r][v] + bval;
            }
        }
        __syncthreads();

        // Fused gate nonlinearity + state update (c in registers).
        float* c_out = c_full + (size_t)t * BATCH * HDIM;
        float* h_out = h_full + (size_t)t * BATCH * HDIM;
#pragma unroll
        for (int e = 0; e < 8; ++e) {
            unsigned idx = (unsigned)e * 256u + tid;   // 64 rows x 32 cols
            unsigned row = idx >> 5;
            unsigned c   = idx & 31u;
            float fz = fsigmoid(smZ[(0u * 64u + row) * 32u + c]);
            float iz = fsigmoid(smZ[(1u * 64u + row) * 32u + c]);
            float gz = tanhf  (smZ[(2u * 64u + row) * 32u + c]);
            float oz = fsigmoid(smZ[(3u * 64u + row) * 32u + c]);
            float ct = creg[e] * fz + iz * gz;
            float ht = oz * tanhf(ct);
            creg[e] = ct;
            size_t gidx = (size_t)row * HDIM + j0 + c;
            c_out[gidx] = ct;
            h_out[gidx] = ht;
            hn[gidx]    = f32_to_bf16_rne(ht);
        }

        // All WGs must finish h_next before anyone starts step t+1.
        grid_barrier(bar, (unsigned)NWG * (unsigned)(t + 1));
    }
}

// ---------------------------------------------------------------------------
extern "C" void kernel_launch(void* const* d_in, const int* in_sizes, int n_in,
                              void* d_out, int out_size, void* d_ws, size_t ws_size,
                              hipStream_t stream)
{
    (void)in_sizes; (void)n_in; (void)out_size; (void)ws_size;
    const float* x   = (const float*)d_in[0];
    const float* c0  = (const float*)d_in[1];
    const float* h0  = (const float*)d_in[2];
    const float* Wf  = (const float*)d_in[3];
    const float* bf_ = (const float*)d_in[4];
    const float* Wi  = (const float*)d_in[5];
    const float* bi_ = (const float*)d_in[6];
    const float* Wg  = (const float*)d_in[7];
    const float* bg_ = (const float*)d_in[8];
    const float* Wo  = (const float*)d_in[9];
    const float* bo_ = (const float*)d_in[10];

    char* ws = (char*)d_ws;
    unsigned short* WxP = (unsigned short*)ws; ws += (size_t)4*64*16*32*16*2;  //  4 MiB
    unsigned short* WhP = (unsigned short*)ws; ws += (size_t)4*64*32*32*16*2;  //  8 MiB
    float* bias         = (float*)ws;          ws += (size_t)4096*4;
    unsigned short* hA  = (unsigned short*)ws; ws += (size_t)BATCH*HDIM*2;
    unsigned short* hB  = (unsigned short*)ws; ws += (size_t)BATCH*HDIM*2;
    unsigned* bar       = (unsigned*)ws;       ws += 256;                      // barrier
    unsigned short* xb  = (unsigned short*)ws; ws += (size_t)L_SEQ*BATCH*EDIM*2; // 32 MiB

    pack_wx_kernel<<<512, 256, 0, stream>>>(Wf, Wi, Wg, Wo, WxP);
    pack_wh_kernel<<<1024, 256, 0, stream>>>(Wf, Wi, Wg, Wo, WhP);
    conv_x_kernel<<<8192, 256, 0, stream>>>(x, xb);
    init_kernel<<<256, 256, 0, stream>>>(h0, bf_, bi_, bg_, bo_, hA, bias, bar);

    float* c_full = (float*)d_out;
    float* h_full = (float*)d_out + (size_t)L_SEQ * BATCH * HDIM;
    lstm_scan_kernel<<<NWG, 256, 0, stream>>>(
        xb, WxP, WhP, bias, c0, hA, hB, c_full, h_full, bar);
}
